// PolyLoRATensor_65094524338793
// MI455X (gfx1250) — compile-verified, matrix operand order checked by reading, not run on hardware
//
#include <hip/hip_runtime.h>

// ---------------------------------------------------------------------------
// MI455X (gfx1250) fused PolyLoRA:  out = X @ W^T + bias + (X @ A) @ B / 16
// Memory-bound (~576 MB HBM @ 23.3 TB/s ~= 25us floor):
//   - single pass over X (rank-16 LoRA fused into the GEMM K-loop)
//   - bf16 WMMA (v_wmma_f32_16x16x32_bf16), fp32 accumulation
//   - register-prefetch software pipeline for global->LDS staging
//   - all WMMAs unconditional (EXEC all-1s, no exec-mask predication)
//   - non-temporal result stores to preserve L2 for X/W reuse
// ---------------------------------------------------------------------------

typedef __attribute__((ext_vector_type(16))) __bf16 v16bf;
typedef __attribute__((ext_vector_type(8)))  __bf16 v8bf;
typedef __attribute__((ext_vector_type(4)))  __bf16 v4bf;
typedef __attribute__((ext_vector_type(8)))  float  v8f;

#define D_MODEL   4096
#define SEQ_ROWS  16384   // 8 * 2048
#define N_SKILLS  8
#define RANKR     16
#define LEAF      8

// ------------------------- small prep kernels ------------------------------

__global__ void k_selector(const float* __restrict__ logits,
                           const int* __restrict__ task_ids,
                           float* __restrict__ mw) {
  int b = threadIdx.x;
  if (b < 8) {
    int t = task_ids[b];
    float p[N_SKILLS];
    float s = 0.f;
#pragma unroll
    for (int i = 0; i < N_SKILLS; ++i) {
      p[i] = 1.f / (1.f + expf(-logits[t * N_SKILLS + i]));
      s += p[i];
    }
    float inv = 1.f / (s + 1e-12f);
#pragma unroll
    for (int i = 0; i < N_SKILLS; ++i) mw[b * N_SKILLS + i] = p[i] * inv;
  }
}

// One block per (tensor a/b, pair 01/23, skill).  Leaf outer product
// [16 r][64], LayerNorm jointly over all 1024 elements.
__global__ void k_factors(const float* __restrict__ wla,
                          const float* __restrict__ wlb,
                          const float* __restrict__ ln1w, const float* __restrict__ ln1b,
                          const float* __restrict__ ln2w, const float* __restrict__ ln2b,
                          float* __restrict__ fac) {
  int bid  = blockIdx.x;
  int t    = bid >> 4;          // 0 = wla, 1 = wlb
  int pair = (bid >> 3) & 1;    // 0 = wl0*wl1, 1 = wl2*wl3
  int s    = bid & 7;
  const float* wl = t ? wlb : wla;
  const float* lw = pair ? ln2w : ln1w;
  const float* lb = pair ? ln2b : ln1b;

  __shared__ float vals[1024];
  __shared__ float rs[256], rq[256];
  int tid = threadIdx.x;
  float sum = 0.f, sq = 0.f;
  for (int e = tid; e < 1024; e += 256) {
    int r = e >> 6, i = (e >> 3) & 7, j = e & 7;
    float a = wl[(((pair * 2 + 0) * 8 + s) * RANKR + r) * LEAF + i];
    float c = wl[(((pair * 2 + 1) * 8 + s) * RANKR + r) * LEAF + j];
    float v = a * c;
    vals[e] = v;
    sum += v;
    sq  += v * v;
  }
  rs[tid] = sum; rq[tid] = sq;
  __syncthreads();
  for (int o = 128; o > 0; o >>= 1) {
    if (tid < o) { rs[tid] += rs[tid + o]; rq[tid] += rq[tid + o]; }
    __syncthreads();
  }
  float mean = rs[0] * (1.f / 1024.f);
  float var  = rq[0] * (1.f / 1024.f) - mean * mean;
  float inv  = rsqrtf(var + 1e-5f);
  float* outp = fac + (t * 2 + pair) * 8192 + s * 1024;
  for (int e = tid; e < 1024; e += 256) {
    int r = e >> 6, ij = e & 63;
    outp[e] = (vals[e] - mean) * inv * lw[r * 64 + ij] + lb[r * 64 + ij];
  }
}

// Per-batch mixture.  Abf[b][r][d] (bf16, k-major) and Blbf[b][r][d] (bf16,
// with 1/RANK folded in).
__global__ void k_mix(const float* __restrict__ mw, const float* __restrict__ fac,
                      __bf16* __restrict__ Abf, __bf16* __restrict__ Blbf) {
  int b = blockIdx.x >> 4, r = blockIdx.x & 15;
  const float* w01a = fac;
  const float* w23a = fac + 8192;
  const float* w01b = fac + 16384;
  const float* w23b = fac + 24576;
  float m[N_SKILLS];
#pragma unroll
  for (int s = 0; s < N_SKILLS; ++s) m[s] = mw[b * N_SKILLS + s];
  for (int d = threadIdx.x; d < D_MODEL; d += 256) {
    int i = d >> 6, j = d & 63;
    float sa = 0.f, sb = 0.f;
#pragma unroll
    for (int s = 0; s < N_SKILLS; ++s) {
      int base = s * 1024 + r * 64;
      sa += m[s] * w01a[base + i] * w23a[base + j];
      sb += m[s] * w01b[base + i] * w23b[base + j];
    }
    size_t o = ((size_t)b * RANKR + r) * D_MODEL + d;
    Abf[o]  = (__bf16)sa;
    Blbf[o] = (__bf16)(sb * (1.f / 16.f));
  }
}

// ------------------------------ main GEMM ----------------------------------
// C[128x128] tile per block over K=4096 (BK=32).  8 waves as 2(M) x 4(N);
// each wave owns 4x2 16x16 accumulators plus a redundant 64x16 h strip
// (one extra WMMA per subtile-row per k-step, unconditional on all waves so
// WMMA always runs with EXEC all-1s).  Epilogue adds h @ B via one more WMMA
// per subtile (K padded 16->32 with zeros) plus bias, then NT-stores.

__device__ __forceinline__ v16bf load_frag(const __bf16* base, int row0, int lane) {
  // 16-bit A/B fragment (ISA 7.12.2): lanes 0-15 row m, K 0-7 & 16-23;
  // lanes 16-31 row m, K 8-15 & 24-31.  LDS tile is [rows][32] bf16.
  int hf = lane >> 4;
  const __bf16* p = base + (row0 + (lane & 15)) * 32 + hf * 8;
  union { v16bf v; v8bf h[2]; } u;
  u.h[0] = *(const v8bf*)(p);        // K  0..7  (or 8..15)
  u.h[1] = *(const v8bf*)(p + 16);   // K 16..23 (or 24..31)
  return u.v;
}

__global__ __launch_bounds__(256)
void k_gemm(const float* __restrict__ x, const float* __restrict__ w,
            const float* __restrict__ bias,
            const __bf16* __restrict__ Abf, const __bf16* __restrict__ Blbf,
            float* __restrict__ out) {
  __shared__ __bf16 Xs[128 * 32];
  __shared__ __bf16 Wsh[128 * 32];
  __shared__ __bf16 As[16 * 32];
  __shared__ __bf16 Hs[128 * 32];    // h tile, K padded to 32 with zeros
  __shared__ __bf16 Bls[128 * 32];   // B_lora^T tile [n][r], padded to 32

  const int tid  = threadIdx.x;
  const int lane = tid & 31;
  const int wave = tid >> 5;
  const int wm   = wave >> 2;        // 0..1
  const int wn   = wave & 3;         // 0..3
  const int hf   = lane >> 4;
  const int n0   = blockIdx.x * 128;
  const int m0   = blockIdx.y * 128;
  const int b    = m0 >> 11;         // 2048 rows per batch

  v8f zero;
#pragma unroll
  for (int q = 0; q < 8; ++q) zero[q] = 0.f;
  v8f acc[4][2];
  v8f hacc[4];
#pragma unroll
  for (int i = 0; i < 4; ++i) {
    acc[i][0] = zero; acc[i][1] = zero; hacc[i] = zero;
  }

  // one-time staging: B_lora tile [n_local][r], r in 0..31 (16..31 -> 0)
  for (int e = 0; e < 16; ++e) {
    int idx = e * 256 + tid;          // 0..4095
    int nl = idx >> 5, r = idx & 31;
    __bf16 v = (__bf16)0.f;
    if (r < 16) v = Blbf[((size_t)b * RANKR + r) * D_MODEL + n0 + nl];
    Bls[nl * 32 + r] = v;
  }
  // zero h tile (pad region stays zero)
  {
    v8bf z8;
#pragma unroll
    for (int q = 0; q < 8; ++q) z8[q] = (__bf16)0.f;
    for (int e = tid; e < 128 * 32 / 8; e += 256) *(v8bf*)&Hs[e * 8] = z8;
  }

  // register-prefetch software pipeline
  float4 xr[4], wr[4];
  v8bf ar;
#pragma unroll
  for (int q = 0; q < 8; ++q) ar[q] = (__bf16)0.f;
  const int srow = tid >> 3;              // 0..31  (staging row group base)
  const int sc4  = tid & 7;               // float4 column within 32-wide K
  const int arr  = tid >> 2, ark = (tid & 3) * 8;

  // issue global loads for k0 = 0
#pragma unroll
  for (int it = 0; it < 4; ++it) {
    int row = it * 32 + srow;
    xr[it] = *(const float4*)(x + (size_t)(m0 + row) * D_MODEL + sc4 * 4);
    wr[it] = *(const float4*)(w + (size_t)(n0 + row) * D_MODEL + sc4 * 4);
  }
  if (tid < 64)
    ar = *(const v8bf*)(Abf + ((size_t)b * RANKR + arr) * D_MODEL + ark);

  for (int k0 = 0; k0 < D_MODEL; k0 += 32) {
    // ---- convert prefetched tile and store to LDS ----
#pragma unroll
    for (int it = 0; it < 4; ++it) {
      int row = it * 32 + srow;
      v4bf xb = { (__bf16)xr[it].x, (__bf16)xr[it].y,
                  (__bf16)xr[it].z, (__bf16)xr[it].w };
      *(v4bf*)&Xs[row * 32 + sc4 * 4] = xb;
      v4bf wb = { (__bf16)wr[it].x, (__bf16)wr[it].y,
                  (__bf16)wr[it].z, (__bf16)wr[it].w };
      *(v4bf*)&Wsh[row * 32 + sc4 * 4] = wb;
    }
    if (tid < 64) *(v8bf*)&As[arr * 32 + ark] = ar;
    __syncthreads();

    // ---- issue next tile's global loads (overlap with WMMA below) ----
    int kn = k0 + 32;
    if (kn < D_MODEL) {
#pragma unroll
      for (int it = 0; it < 4; ++it) {
        int row = it * 32 + srow;
        xr[it] = *(const float4*)(x + (size_t)(m0 + row) * D_MODEL + kn + sc4 * 4);
        wr[it] = *(const float4*)(w + (size_t)(n0 + row) * D_MODEL + kn + sc4 * 4);
      }
      if (tid < 64)
        ar = *(const v8bf*)(Abf + ((size_t)b * RANKR + arr) * D_MODEL + kn + ark);
    }

    // ---- compute: 12 WMMAs/wave, all unconditional ----
    v16bf bf0 = load_frag(Wsh, wn * 32 + 0, lane);
    v16bf bf1 = load_frag(Wsh, wn * 32 + 16, lane);
    v16bf al  = load_frag(As, 0, lane);
#pragma unroll
    for (int i = 0; i < 4; ++i) {
      v16bf af = load_frag(Xs, wm * 64 + i * 16, lane);
      acc[i][0] = __builtin_amdgcn_wmma_f32_16x16x32_bf16(
          false, af, false, bf0, (short)0, acc[i][0], false, false);
      acc[i][1] = __builtin_amdgcn_wmma_f32_16x16x32_bf16(
          false, af, false, bf1, (short)0, acc[i][1], false, false);
      hacc[i] = __builtin_amdgcn_wmma_f32_16x16x32_bf16(
          false, af, false, al, (short)0, hacc[i], false, false);
    }
    __syncthreads();
  }

  // ---- write h to LDS (bf16); waves with same wm hold identical h, so only
  //      the wn==0 column stores (post-loop, not perf-critical) ----
  if (wn == 0) {
#pragma unroll
    for (int i = 0; i < 4; ++i) {
#pragma unroll
      for (int e = 0; e < 8; ++e) {
        int ml = wm * 64 + i * 16 + e + hf * 8;   // C layout: VGPR e -> M=e / e+8
        Hs[ml * 32 + (lane & 15)] = (__bf16)hacc[i][e];
      }
    }
  }
  __syncthreads();

  // ---- adapter epilogue: acc += h(16x32,pad) @ Bl(32x16) ----
  v16bf bb0 = load_frag(Bls, wn * 32 + 0, lane);
  v16bf bb1 = load_frag(Bls, wn * 32 + 16, lane);
#pragma unroll
  for (int i = 0; i < 4; ++i) {
    v16bf ah = load_frag(Hs, wm * 64 + i * 16, lane);
    acc[i][0] = __builtin_amdgcn_wmma_f32_16x16x32_bf16(
        false, ah, false, bb0, (short)0, acc[i][0], false, false);
    acc[i][1] = __builtin_amdgcn_wmma_f32_16x16x32_bf16(
        false, ah, false, bb1, (short)0, acc[i][1], false, false);
  }

  // ---- bias + non-temporal store (write-once stream; keep L2 for X/W) ----
#pragma unroll
  for (int j = 0; j < 2; ++j) {
    int n = n0 + wn * 32 + j * 16 + (lane & 15);
    float bv = bias[n];
#pragma unroll
    for (int i = 0; i < 4; ++i) {
      int mbase = m0 + wm * 64 + i * 16;
#pragma unroll
      for (int e = 0; e < 8; ++e) {
        int m = mbase + e + hf * 8;
        __builtin_nontemporal_store(acc[i][j][e] + bv,
                                    out + (size_t)m * D_MODEL + n);
      }
    }
  }
}

// ------------------------------- launcher ----------------------------------

extern "C" void kernel_launch(void* const* d_in, const int* in_sizes, int n_in,
                              void* d_out, int out_size, void* d_ws, size_t ws_size,
                              hipStream_t stream) {
  const float* x             = (const float*)d_in[0];
  const int*   task_ids      = (const int*)d_in[1];
  const float* weight        = (const float*)d_in[2];
  const float* bias          = (const float*)d_in[3];
  const float* module_logits = (const float*)d_in[4];
  const float* wla           = (const float*)d_in[5];
  const float* wlb           = (const float*)d_in[6];
  const float* ln1w          = (const float*)d_in[7];
  const float* ln1b          = (const float*)d_in[8];
  const float* ln2w          = (const float*)d_in[9];
  const float* ln2b          = (const float*)d_in[10];
  float* out = (float*)d_out;

  char* ws = (char*)d_ws;
  float*  mw   = (float*)ws;                       // 64 f32
  float*  fac  = (float*)(ws + 256);               // 4 * 8192 f32 = 128 KB
  __bf16* Abf  = (__bf16*)(ws + 131328);           // 8*16*4096 bf16 = 1 MB
  __bf16* Blbf = (__bf16*)(ws + 131328 + 1048576); // 1 MB

  k_selector<<<1, 32, 0, stream>>>(module_logits, task_ids, mw);
  k_factors<<<32, 256, 0, stream>>>(wla, wlb, ln1w, ln1b, ln2w, ln2b, fac);
  k_mix<<<128, 256, 0, stream>>>(mw, fac, Abf, Blbf);

  dim3 grid(D_MODEL / 128, SEQ_ROWS / 128);        // 32 x 128
  k_gemm<<<grid, 256, 0, stream>>>(x, weight, bias, Abf, Blbf, out);
}
